// DeformableSparseAttn3D_67233418052174
// MI455X (gfx1250) — matrix operand order repaired and checked by reading, not compiled, not run on hardware
//
#include <hip/hip_runtime.h>

// DeformableSparseAttn3D for gfx1250 (MI455X, wave32, WMMA).
// All GEMM-shaped stages -> v_wmma_f32_16x16x32_f16 (f16 in, f32 acc).
// Weights pre-transposed to f16 W^T (NxK row-major): B-frag = one contiguous
// 32B load per lane. Activations row-major f16, K padded to x32: A-frag = two
// contiguous 16B loads. The 3x3 conv (dominant ~36 GFLOP) is an implicit GEMM
// over a halo-padded channel-last f16 image; both the conv and the generic
// GEMM use an explicit 2-stage ping/pong software pipeline (loads of stage k+1
// issue before the WMMAs of stage k) so memory latency hides under the matrix
// pipe. Irregular ops (topk, LN, softmax, bilinear) run on VALU.

#define C_ 128
#define H_ 4
#define FH_ 96
#define FW_ 320
#define P_ (FH_*FW_)     // 30720 pixels
#define PW_ (FW_+2)      // padded width 322
#define PH_ (FH_+2)      // padded height 98

typedef __attribute__((ext_vector_type(16))) _Float16 v16h;
typedef __attribute__((ext_vector_type(8)))  _Float16 v8h;
typedef __attribute__((ext_vector_type(8)))  float    v8f;

__device__ __forceinline__ float gelu_f(float x){ return 0.5f*x*(1.0f+erff(x*0.7071067811865475f)); }
__device__ __forceinline__ float lrelu_f(float x){ return x>0.f ? x : 0.01f*x; }
__device__ __forceinline__ float wred_sum(float v){ for(int o=16;o;o>>=1) v += __shfl_xor(v,o,32); return v; }
__device__ __forceinline__ float wred_max(float v){ for(int o=16;o;o>>=1) v = fmaxf(v,__shfl_xor(v,o,32)); return v; }

// ---------------- weight prep ----------------
// fp32 W (K x N) row-major  ->  f16 W^T (Np x Kp) row-major, zero-padded
__global__ void wtrans_kernel(const float* __restrict__ W, int K, int Nn,
                              _Float16* __restrict__ Wt, int Kp, long total){
  long idx = (long)blockIdx.x*blockDim.x + threadIdx.x;
  if (idx >= total) return;
  int n = (int)(idx / Kp), k = (int)(idx % Kp);
  float v = 0.f;
  if (k < K && n < Nn) v = W[(long)k*Nn + n];
  Wt[idx] = (_Float16)v;
}

__global__ void padbias_kernel(const float* __restrict__ in, float* __restrict__ out, int n, int np){
  int i = blockIdx.x*blockDim.x + threadIdx.x;
  if (i >= np) return;
  out[i] = (i < n) ? in[i] : 0.f;
}

// image (C,FH,FW) f32 -> halo-padded channel-last (PH_,PW_,C) f16, border = 0
__global__ void xprep_pad_kernel(const float* __restrict__ x, _Float16* __restrict__ xp){
  long idx = (long)blockIdx.x*blockDim.x + threadIdx.x;
  if (idx >= (long)PH_*PW_*C_) return;
  int c = (int)(idx & 127);
  long pp = idx >> 7;
  int px = (int)(pp % PW_), py = (int)(pp / PW_);
  int xx = px - 1, yy = py - 1;
  float v = 0.f;
  if (xx >= 0 && xx < FW_ && yy >= 0 && yy < FH_) v = x[c*P_ + yy*FW_ + xx];
  xp[idx] = (_Float16)v;
}

// ---------------- generic WMMA GEMM (NT n-tiles/wave, 2-stage pipeline) ----------------
// D(m,n) = EPI( scale * sum_k A[m,k]*Bt[n,k] + bias[n] )
template<int EPI, bool HB, int NT>   // EPI: 0 none, 1 gelu, 2 lrelu
__global__ void gemm_kernel(const _Float16* __restrict__ A, int lda,
                            const _Float16* __restrict__ Bt, int ldb,
                            const float* __restrict__ bias, float scale,
                            float* __restrict__ D, int ldd,
                            _Float16* __restrict__ D16, int ldd16,
                            int Mt, int Nt, int Kt, int Mmax)
{
  int wave = (int)(((long)blockIdx.x*blockDim.x + threadIdx.x) >> 5);
  int lane = threadIdx.x & 31;
  int Ng = Nt / NT;
  if (wave >= Mt*Ng) return;
  int mt = wave / Ng, n0 = (wave % Ng) * NT;
  int r = lane & 15, hi = lane >> 4;
  const _Float16* pa = A  + (long)(mt*16 + r)*lda + hi*16;
  const _Float16* pb = Bt + (long)(n0*16 + r)*ldb + hi*16;
  v8f acc[NT];
  #pragma unroll
  for (int t=0;t<NT;++t) for (int e=0;e<8;++e) acc[t][e]=0.f;

  union U { v16h v; v8h h2[2]; };
  U a0, a1;
  v16h b0[NT], b1[NT];
  // prologue: stage 0
  a0.h2[0] = *(const v8h*)pa;            // K = hi*16 + (0..7)
  a0.h2[1] = *(const v8h*)(pa + 8);      // K = hi*16 + 8 + (0..7)
  #pragma unroll
  for (int t=0;t<NT;++t) b0[t] = *(const v16h*)(pb + (long)t*16*ldb);

  int kt = 0;
  for (; kt+2 <= Kt; kt += 2){
    // issue loads for stage kt+1 (distinct regs -> overlap with WMMAs below)
    a1.h2[0] = *(const v8h*)(pa + 32);
    a1.h2[1] = *(const v8h*)(pa + 40);
    #pragma unroll
    for (int t=0;t<NT;++t) b1[t] = *(const v16h*)(pb + (long)t*16*ldb + 32);
    // WMMA group for stage kt (one stage of load slack)
    #pragma unroll
    for (int t=0;t<NT;++t)
      acc[t] = __builtin_amdgcn_wmma_f32_16x16x32_f16(false, a0.v, false, b0[t], (short)0, acc[t], false, false);
    // issue loads for stage kt+2
    if (kt+2 < Kt){
      a0.h2[0] = *(const v8h*)(pa + 64);
      a0.h2[1] = *(const v8h*)(pa + 72);
      #pragma unroll
      for (int t=0;t<NT;++t) b0[t] = *(const v16h*)(pb + (long)t*16*ldb + 64);
    }
    // WMMA group for stage kt+1
    #pragma unroll
    for (int t=0;t<NT;++t)
      acc[t] = __builtin_amdgcn_wmma_f32_16x16x32_f16(false, a1.v, false, b1[t], (short)0, acc[t], false, false);
    pa += 64; pb += 64;
  }
  if (kt < Kt){   // odd-Kt tail: stage already resident in a0/b0
    #pragma unroll
    for (int t=0;t<NT;++t)
      acc[t] = __builtin_amdgcn_wmma_f32_16x16x32_f16(false, a0.v, false, b0[t], (short)0, acc[t], false, false);
  }

  bool full = (mt*16 + 16) <= Mmax;
  #pragma unroll
  for (int t=0;t<NT;++t){
    int n = (n0+t)*16 + r;
    float bb = HB ? bias[n] : 0.f;
    #pragma unroll
    for (int e=0;e<8;++e){
      int m = mt*16 + e + hi*8;
      if (!full && m >= Mmax) continue;
      float v = acc[t][e]*scale + bb;
      if (EPI==1) v = gelu_f(v);
      else if (EPI==2) v = lrelu_f(v);
      if (D)   D  [(long)m*ldd   + n] = v;
      if (D16) D16[(long)m*ldd16 + n] = (_Float16)v;
    }
  }
}

// ---------------- implicit-GEMM 3x3 conv, fused bias + residual ----------------
// feats[(h*C+c), p] = conv(x)[h*C+c, p] + conv_b[h*C+c] + x[c, p]
// Flat k-step kt in [0,36): s=kt/4 -> (ky,kx)=(s/3,s%3), ct=kt&3 -> channels ct*32..+31.
// Fully unrolled with an explicit 2-stage ping/pong pipeline: all addresses fold
// to immediates off two base pointers; loads of step k+1 issue before WMMAs of k.
__global__ __launch_bounds__(256) void conv_kernel(const _Float16* __restrict__ xp,
     const _Float16* __restrict__ Wt, const float* __restrict__ bias,
     const float* __restrict__ x32, float* __restrict__ feats){
  int pt   = blockIdx.x;          // 16-pixel tile (320%16==0 -> single image row)
  int wv   = threadIdx.x >> 5;    // 8 waves/block x 4 n-tiles = all 512 outputs
  int lane = threadIdx.x & 31;
  int r = lane & 15, hi = lane >> 4;
  int p = pt*16 + r;
  int y = p / FW_, xq = p % FW_;
  v8f acc[4];
  for (int t=0;t<4;++t) for (int e=0;e<8;++e) acc[t][e]=0.f;
  const _Float16* wbase = Wt + (long)(wv*64 + r)*1152 + hi*16;    // + t*16*1152 + kt*32 (imm)
  const _Float16* pbase = xp + (((long)(y*PW_ + xq))<<7) + hi*16; // + ((ky*PW_+kx)<<7)+ct*32 (imm)

  union U { v16h v; v8h h2[2]; };
  #define AOFF(KT) ((((long)(((KT)/4)/3)*PW_ + (((KT)/4)%3))<<7) + ((KT)&3)*32)
  #define LDA(dst, KT) { dst.h2[0] = *(const v8h*)(pbase + AOFF(KT)); \
                         dst.h2[1] = *(const v8h*)(pbase + AOFF(KT) + 8); }
  #define LDB(dst, KT) { _Pragma("unroll") \
                         for (int t_=0;t_<4;++t_) dst[t_] = *(const v16h*)(wbase + (long)t_*16*1152 + (KT)*32); }
  #define WM(av, bv)   { _Pragma("unroll") \
                         for (int t_=0;t_<4;++t_) \
                           acc[t_] = __builtin_amdgcn_wmma_f32_16x16x32_f16(false, av.v, false, bv[t_], (short)0, acc[t_], false, false); }
  U a0, a1;
  v16h b0[4], b1[4];
  LDA(a0, 0); LDB(b0, 0);                 // prologue: stage 0
  #pragma unroll
  for (int kt=0; kt<36; kt+=2){
    LDA(a1, kt+1); LDB(b1, kt+1);         // stage kt+1 loads before stage-kt WMMAs
    WM(a0, b0);
    if (kt+2 < 36){ LDA(a0, kt+2); LDB(b0, kt+2); }
    WM(a1, b1);
  }
  #undef AOFF
  #undef LDA
  #undef LDB
  #undef WM

  int pbase_o = pt*16 + hi*8;
  for (int t=0;t<4;++t){
    int n = (wv*4+t)*16 + r;
    int c = n & 127;
    float bb = bias[n];
    const float* xr   = x32   + (long)c*P_ + pbase_o;
    float*       orow = feats + (long)n*P_ + pbase_o;
    for (int e=0;e<8;++e) orow[e] = acc[t][e] + bb + xr[e];  // contiguous 32B store
  }
}

// ---------------- small / irregular stages (VALU) ----------------
__global__ void gather_kernel(const float* __restrict__ eh, const float* __restrict__ ew,
                              const float* __restrict__ ed, const int* __restrict__ idxs,
                              float* hb, float* wb, float* db, _Float16* pos16, int M, int Mp){
  int t = blockIdx.x*blockDim.x + threadIdx.x;
  if (t >= Mp*C_) return;
  int m = t >> 7, c = t & 127;
  float h=0.f,w=0.f,d=0.f;
  if (m < M){
    h = eh[idxs[m]*C_ + c];
    w = ew[idxs[M+m]*C_ + c];
    d = ed[idxs[2*M+m]*C_ + c];
  }
  hb[t]=h; wb[t]=w; db[t]=d;
  long pb = (long)m*384;
  pos16[pb+c]=(_Float16)h; pos16[pb+128+c]=(_Float16)w; pos16[pb+256+c]=(_Float16)d;
}

__global__ void ln_kernel(const float* __restrict__ in, const float* __restrict__ g,
                          const float* __restrict__ b, float* __restrict__ out,
                          _Float16* __restrict__ out16, int M, int Mp){
  int wave = (int)(((long)blockIdx.x*blockDim.x + threadIdx.x) >> 5);
  int lane = threadIdx.x & 31;
  if (wave >= Mp) return;
  long rb = (long)wave*C_;
  if (wave >= M){
    for (int e=0;e<4;++e){
      out[rb + lane*4 + e] = 0.f;
      if (out16) out16[rb + lane*4 + e] = (_Float16)0.f;
    }
    return;
  }
  float v[4]; float s = 0.f;
  for (int e=0;e<4;++e){ v[e] = in[rb+lane*4+e]; s += v[e]; }
  float mean = wred_sum(s) * (1.f/128.f);
  float vs = 0.f;
  for (int e=0;e<4;++e){ float d = v[e]-mean; vs += d*d; }
  float inv = rsqrtf(wred_sum(vs)*(1.f/128.f) + 1e-5f);
  for (int e=0;e<4;++e){
    int c = lane*4+e;
    float o = (v[e]-mean)*inv*g[c] + b[c];
    out[rb+c] = o;
    if (out16) out16[rb+c] = (_Float16)o;
  }
}

__global__ void fo_kernel(const float* __restrict__ comb, const float* __restrict__ hb,
                          const float* __restrict__ wb, const float* __restrict__ db,
                          _Float16* __restrict__ fo16, int Mp){
  int wave = (int)(((long)blockIdx.x*blockDim.x + threadIdx.x) >> 5);
  int lane = threadIdx.x & 31;
  if (wave >= Mp) return;
  long rb = (long)wave*C_;
  float c4[4];
  for (int e=0;e<4;++e) c4[e] = comb[rb+lane*4+e];
  const float* parts[3] = {hb, wb, db};
  #pragma unroll
  for (int pz=0; pz<3; ++pz){
    float t[4]; float ss=0.f;
    for (int e=0;e<4;++e){ float x = parts[pz][rb+lane*4+e]; t[e] = c4[e]*x + x; ss += t[e]*t[e]; }
    ss = wred_sum(ss);
    float inv = 1.f / fmaxf(sqrtf(ss), 1e-12f);
    for (int e=0;e<4;++e) fo16[(long)wave*384 + pz*C_ + lane*4 + e] = (_Float16)(t[e]*inv);
  }
}

// one wave per ref point: per-lane top-16 insertion + 16 wave-wide min-selection rounds
__global__ void topk_kernel(const float* __restrict__ selref, const float* __restrict__ qpos,
                            int N, int M, int* __restrict__ nei){
  int wave = (int)(((long)blockIdx.x*blockDim.x + threadIdx.x) >> 5);
  int lane = threadIdx.x & 31;
  if (wave >= N) return;
  float rx = selref[wave*3+0], ry = selref[wave*3+1], rz = selref[wave*3+2];
  float d[16]; int id[16];
  for (int t=0;t<16;++t){ d[t] = 3.0e38f; id[t] = 0x7fffffff; }
  for (int m=lane; m<M; m+=32){
    float dx = rx - qpos[m*3+0];
    float dy = ry - qpos[m*3+1];
    float dz = rz - qpos[m*3+2];
    float dd = dx*dx + dy*dy + dz*dz;
    if (dd < d[15]){
      int j = 15;
      while (j > 0 && d[j-1] > dd){ d[j]=d[j-1]; id[j]=id[j-1]; --j; }
      d[j] = dd; id[j] = m;
    }
  }
  int ptr = 0;
  for (int rr=0; rr<16; ++rr){
    float bd = (ptr<16)? d[ptr] : 3.0e38f;
    int   bi = (ptr<16)? id[ptr] : 0x7fffffff;
    for (int o=16;o;o>>=1){
      float od = __shfl_xor(bd, o, 32);
      int   oi = __shfl_xor(bi, o, 32);
      if (od < bd || (od==bd && oi<bi)){ bd=od; bi=oi; }
    }
    if (ptr<16 && d[ptr]==bd && id[ptr]==bi) ++ptr;
    if (lane==0) nei[wave*16 + rr] = bi;
  }
}

// per ref point: tiny weight-MLP (3->16->16) + pcv = g^T @ wts / 16, packed (c*16+w)
__global__ void wts_pcv_kernel(const float* __restrict__ qf, const int* __restrict__ nei,
                               const float* __restrict__ qpos, const float* __restrict__ selref,
                               const float* __restrict__ wn1, const float* __restrict__ wnb1,
                               const float* __restrict__ wn2, const float* __restrict__ wnb2,
                               _Float16* __restrict__ pcv, int N){
  __shared__ float sg[16][128];
  __shared__ float sw[16][16];
  int i = blockIdx.x;
  int tid = threadIdx.x;  // 128
  long ob = (long)i*2048;
  if (i >= N){
    for (int w=0;w<16;++w) pcv[ob + tid*16 + w] = (_Float16)0.f;
    return;
  }
  for (int k=0;k<16;++k) sg[k][tid] = qf[(long)nei[i*16+k]*C_ + tid];
  if (tid < 16){
    int k = tid;
    int q = nei[i*16+k];
    float rx = qpos[q*3+0]-selref[i*3+0];
    float ry = qpos[q*3+1]-selref[i*3+1];
    float rz = qpos[q*3+2]-selref[i*3+2];
    float hd[16];
    for (int w=0;w<16;++w)
      hd[w] = gelu_f(rx*wn1[w] + ry*wn1[16+w] + rz*wn1[32+w] + wnb1[w]);
    for (int w=0;w<16;++w){
      float s = wnb2[w];
      for (int j=0;j<16;++j) s += hd[j]*wn2[j*16+w];
      sw[k][w] = gelu_f(s);
    }
  }
  __syncthreads();
  int c = tid;
  for (int w=0;w<16;++w){
    float s = 0.f;
    for (int k=0;k<16;++k) s += sg[k][c]*sw[k][w];
    pcv[ob + c*16 + w] = (_Float16)(s * (1.f/16.f));
  }
}

// bilinear sample feats at deformed ref points -> fts (h, i, c) f16
__global__ void sample_kernel(const float* __restrict__ feats, const float* __restrict__ refpts,
                              const float* __restrict__ off, _Float16* __restrict__ fts16,
                              int N, int NP){
  int i = blockIdx.x, h = blockIdx.y, c = threadIdx.x;
  long orow = ((long)h*NP + i)*C_ + c;
  if (i >= N){ fts16[orow] = (_Float16)0.f; return; }
  float nx = refpts[i*2+0] + off[(long)i*16 + h*2 + 0];
  float ny = refpts[i*2+1] + off[(long)i*16 + h*2 + 1];
  nx = fminf(fmaxf(nx, 0.f), 1.f);
  ny = fminf(fmaxf(ny, 0.f), 1.f);
  float gx = (nx+1.f)*0.5f*(float)(FW_-1);
  float gy = (ny+1.f)*0.5f*(float)(FH_-1);
  float x0 = fminf(fmaxf(floorf(gx), 0.f), (float)(FW_-1));
  float y0 = fminf(fmaxf(floorf(gy), 0.f), (float)(FH_-1));
  int x0i=(int)x0, y0i=(int)y0;
  int x1i=min(x0i+1, FW_-1), y1i=min(y0i+1, FH_-1);
  float wx = gx-x0, wy = gy-y0;
  const float* row = feats + ((long)(h*C_+c))*P_;
  float s00=row[y0i*FW_+x0i], s01=row[y0i*FW_+x1i];
  float s10=row[y1i*FW_+x0i], s11=row[y1i*FW_+x1i];
  float s = s00*(1.f-wx)*(1.f-wy) + s01*wx*(1.f-wy) + s10*(1.f-wx)*wy + s11*wx*wy;
  fts16[orow] = (_Float16)s;
}

// wave-per-row softmax over valid columns; padded cols -> 0 so the attn@V GEMM
// can run over the zero-padded K dimension
__global__ void softmax_kernel(const float* __restrict__ S, _Float16* __restrict__ A16,
                               int Mrows, int N, int NP){
  int wave = (int)(((long)blockIdx.x*blockDim.x + threadIdx.x) >> 5);
  int lane = threadIdx.x & 31;
  if (wave >= Mrows) return;
  const float* row = S + (long)wave*NP;
  float mx = -3.0e38f;
  for (int j=lane; j<N; j+=32) mx = fmaxf(mx, row[j]);
  mx = wred_max(mx);
  float s = 0.f;
  for (int j=lane; j<N; j+=32) s += expf(row[j]-mx);
  s = wred_sum(s);
  float inv = 1.f/s;
  _Float16* orow = A16 + (long)wave*NP;
  for (int j=lane; j<NP; j+=32)
    orow[j] = (j<N) ? (_Float16)(expf(row[j]-mx)*inv) : (_Float16)0.f;
}

// v_h (NP x 128) f16 -> v^T (128 x NP) f16 so it can be a B^T operand
__global__ void vtrans_kernel(const _Float16* __restrict__ vin, _Float16* __restrict__ vt, int NP){
  long idx = (long)blockIdx.x*blockDim.x + threadIdx.x;
  if (idx >= (long)C_*NP) return;
  int c = (int)(idx / NP), j = (int)(idx % NP);
  vt[idx] = vin[(long)j*C_ + c];
}

// ho (h, m, c) -> hoR (m, h*128+c)
__global__ void hor_kernel(const _Float16* __restrict__ ho, _Float16* __restrict__ hoR, int Mp){
  long idx = (long)blockIdx.x*blockDim.x + threadIdx.x;
  if (idx >= (long)Mp*512) return;
  int m = (int)(idx / 512), rest = (int)(idx % 512);
  int h = rest >> 7, c = rest & 127;
  hoR[idx] = ho[((long)h*Mp + m)*C_ + c];
}

// ---------------- host orchestration ----------------
// Input order: params dict leaves in insertion order, then the remaining inputs.
enum {
  I_EMB_H=0, I_EMB_W, I_EMB_D,
  I_PE_L1, I_PE_LB1, I_PE_L2, I_PE_LB2, I_PE_LN_G, I_PE_LN_B,
  I_PE_O1, I_PE_OB1, I_PE_O2, I_PE_OB2, I_PE_OLN_G, I_PE_OLN_B,
  I_WN1, I_WNB1, I_WN2, I_WNB2,
  I_OF1, I_OFB1, I_OF2, I_OFB2,
  I_CONV_W, I_CONV_B,
  I_Q1, I_QB1, I_Q2, I_QB2,
  I_K1, I_KB1, I_K2, I_KB2,
  I_V1, I_VB1, I_V2, I_VB2,
  I_PO1, I_POB1, I_PO2, I_POB2,
  I_F1, I_FB1, I_F2, I_FB2,
  I_IMG, I_IDX, I_REFPTS, I_SELREF, I_QPOS
};

extern "C" void kernel_launch(void* const* d_in, const int* in_sizes, int n_in,
                              void* d_out, int out_size, void* d_ws, size_t ws_size,
                              hipStream_t stream)
{
  (void)n_in; (void)out_size; (void)ws_size;
  const int M  = in_sizes[I_QPOS] / 3;
  const int N  = in_sizes[I_REFPTS] / 2;
  const int Mp = (M + 15) & ~15;     // query rows padded to WMMA tile
  const int NP = (N + 31) & ~31;     // ref rows padded (also a K dim -> x32)

  size_t cur = 0;
  auto alloc = [&](size_t bytes)->void*{
    void* p = (char*)d_ws + cur;
    cur += (bytes + 255) & ~(size_t)255;
    return p;
  };
  auto f32in = [&](int i){ return (const float*)d_in[i]; };

  // --- weight prep: fp32 (K,N) -> f16 W^T (Np,Kp), zero-padded ---
  auto WT = [&](int idx, int K, int Nn, int Kp, int Np)->_Float16*{
    _Float16* w = (_Float16*)alloc((size_t)Kp*Np*sizeof(_Float16));
    long total = (long)Kp*Np;
    wtrans_kernel<<<(unsigned)((total+255)/256),256,0,stream>>>(f32in(idx), K, Nn, w, Kp, total);
    return w;
  };
  _Float16* Wt_pel1 = WT(I_PE_L1, 384, 256, 384, 256);
  _Float16* Wt_pel2 = WT(I_PE_L2, 256, 128, 256, 128);
  _Float16* Wt_peo1 = WT(I_PE_O1, 384, 256, 384, 256);
  _Float16* Wt_peo2 = WT(I_PE_O2, 256, 128, 256, 128);
  _Float16* Wt_of1  = WT(I_OF1, 2048, 128, 2048, 128);
  _Float16* Wt_of2  = WT(I_OF2,  128,   8,  128,  16);
  _Float16* Wt_conv = WT(I_CONV_W, 1152, 512, 1152, 512);
  _Float16* Wt_q1 = WT(I_Q1,128,128,128,128);
  _Float16* Wt_q2 = WT(I_Q2,128,128,128,128);
  _Float16* Wt_k1 = WT(I_K1,128,128,128,128);
  _Float16* Wt_k2 = WT(I_K2,128,128,128,128);
  _Float16* Wt_v1 = WT(I_V1,128,128,128,128);
  _Float16* Wt_v2 = WT(I_V2,128,128,128,128);
  _Float16* Wt_po1 = WT(I_PO1,128,128,128,128);
  _Float16* Wt_po2 = WT(I_PO2,128,128,128,128);
  _Float16* Wt_f1 = WT(I_F1, 512, 128, 512, 128);
  _Float16* Wt_f2 = WT(I_F2, 128, 128, 128, 128);
  float* ofb2p = (float*)alloc(16*sizeof(float));
  padbias_kernel<<<1,16,0,stream>>>(f32in(I_OFB2), ofb2p, 8, 16);

  // GEMM dispatch: NT = max n-tiles/wave that divides Ntiles (A reuse)
  auto GEMM = [&](int epi, bool hb, const _Float16* A, int lda, const _Float16* Bt, int ldb,
                  const float* bias, float scale, float* D, int ldd, _Float16* D16, int ldd16,
                  int Mt, int Nt, int Kt, int Mmax){
    int NT = (Nt % 4 == 0) ? 4 : ((Nt % 2 == 0) ? 2 : 1);
    unsigned blocks = (unsigned)(((long)Mt*(Nt/NT) + 3)/4);
    #define GL(E,B,T) gemm_kernel<E,B,T><<<blocks,128,0,stream>>>(A,lda,Bt,ldb,bias,scale,D,ldd,D16,ldd16,Mt,Nt,Kt,Mmax)
    if (epi==0 && !hb){ if (NT==4) GL(0,false,4); else if (NT==2) GL(0,false,2); else GL(0,false,1); }
    else if (epi==0)  { if (NT==4) GL(0,true ,4); else if (NT==2) GL(0,true ,2); else GL(0,true ,1); }
    else if (epi==1)  { if (NT==4) GL(1,true ,4); else if (NT==2) GL(1,true ,2); else GL(1,true ,1); }
    else              { if (NT==4) GL(2,true ,4); else if (NT==2) GL(2,true ,2); else GL(2,true ,1); }
    #undef GL
  };

  // --- stage A: positional encoding of queries ---
  float* hb = (float*)alloc((size_t)Mp*128*4);
  float* wbuf = (float*)alloc((size_t)Mp*128*4);
  float* db = (float*)alloc((size_t)Mp*128*4);
  _Float16* pos16 = (_Float16*)alloc((size_t)Mp*384*2);
  gather_kernel<<<(Mp*128+255)/256,256,0,stream>>>(f32in(I_EMB_H), f32in(I_EMB_W), f32in(I_EMB_D),
      (const int*)d_in[I_IDX], hb, wbuf, db, pos16, M, Mp);

  _Float16* t1 = (_Float16*)alloc((size_t)Mp*256*2);
  GEMM(1,true, pos16,384, Wt_pel1,384, f32in(I_PE_LB1), 1.f, nullptr,0, t1,256, Mp/16,16,12, Mp);
  float* t2 = (float*)alloc((size_t)Mp*128*4);
  GEMM(1,true, t1,256, Wt_pel2,256, f32in(I_PE_LB2), 1.f, t2,128, nullptr,0, Mp/16,8,8, Mp);
  float* comb = (float*)alloc((size_t)Mp*128*4);
  ln_kernel<<<(Mp*32+255)/256,256,0,stream>>>(t2, f32in(I_PE_LN_G), f32in(I_PE_LN_B), comb, nullptr, M, Mp);
  _Float16* fo16 = (_Float16*)alloc((size_t)Mp*384*2);
  fo_kernel<<<(Mp*32+255)/256,256,0,stream>>>(comb, hb, wbuf, db, fo16, Mp);
  _Float16* t3 = (_Float16*)alloc((size_t)Mp*256*2);
  GEMM(1,true, fo16,384, Wt_peo1,384, f32in(I_PE_OB1), 1.f, nullptr,0, t3,256, Mp/16,16,12, Mp);
  float* t4 = (float*)alloc((size_t)Mp*128*4);
  GEMM(1,true, t3,256, Wt_peo2,256, f32in(I_PE_OB2), 1.f, t4,128, nullptr,0, Mp/16,8,8, Mp);
  float* qf = (float*)alloc((size_t)Mp*128*4);
  _Float16* qf16 = (_Float16*)alloc((size_t)Mp*128*2);
  ln_kernel<<<(Mp*32+255)/256,256,0,stream>>>(t4, f32in(I_PE_OLN_G), f32in(I_PE_OLN_B), qf, qf16, M, Mp);

  // --- stage B: kNN neighbours, weight net, pcv, offsets ---
  int* nei = (int*)alloc((size_t)NP*16*4);
  topk_kernel<<<((long)N*32+255)/256,256,0,stream>>>(f32in(I_SELREF), f32in(I_QPOS), N, M, nei);
  _Float16* pcv = (_Float16*)alloc((size_t)NP*2048*2);
  wts_pcv_kernel<<<NP,128,0,stream>>>(qf, nei, f32in(I_QPOS), f32in(I_SELREF),
      f32in(I_WN1), f32in(I_WNB1), f32in(I_WN2), f32in(I_WNB2), pcv, N);
  _Float16* o1 = (_Float16*)alloc((size_t)NP*128*2);
  GEMM(1,true, pcv,2048, Wt_of1,2048, f32in(I_OFB1), 1.f, nullptr,0, o1,128, NP/16,8,64, NP);
  float* offb = (float*)alloc((size_t)NP*16*4);
  GEMM(0,true, o1,128, Wt_of2,128, ofb2p, 1.f, offb,16, nullptr,0, NP/16,1,4, NP);

  // --- stage C: 3x3 conv (implicit GEMM) + residual, then bilinear sampling ---
  _Float16* xp = (_Float16*)alloc((size_t)PH_*PW_*C_*2);
  xprep_pad_kernel<<<(unsigned)(((long)PH_*PW_*C_+255)/256),256,0,stream>>>(f32in(I_IMG), xp);
  float* feats = (float*)alloc((size_t)512*P_*4);
  conv_kernel<<<P_/16,256,0,stream>>>(xp, Wt_conv, f32in(I_CONV_B), f32in(I_IMG), feats);
  _Float16* fts16 = (_Float16*)alloc((size_t)4*NP*128*2);
  dim3 sgrid((unsigned)NP, 4);
  sample_kernel<<<sgrid,128,0,stream>>>(feats, f32in(I_REFPTS), offb, fts16, N, NP);

  // --- stage D: q / k / v projections ---
  int HN = 4*NP;
  _Float16* ka = (_Float16*)alloc((size_t)HN*128*2);
  _Float16* kk = (_Float16*)alloc((size_t)HN*128*2);
  _Float16* va = (_Float16*)alloc((size_t)HN*128*2);
  _Float16* vv = (_Float16*)alloc((size_t)HN*128*2);
  GEMM(2,true, fts16,128, Wt_k1,128, f32in(I_KB1), 1.f, nullptr,0, ka,128, HN/16,8,4, HN);
  GEMM(1,true, ka,128,    Wt_k2,128, f32in(I_KB2), 1.f, nullptr,0, kk,128, HN/16,8,4, HN);
  GEMM(2,true, fts16,128, Wt_v1,128, f32in(I_VB1), 1.f, nullptr,0, va,128, HN/16,8,4, HN);
  GEMM(1,true, va,128,    Wt_v2,128, f32in(I_VB2), 1.f, nullptr,0, vv,128, HN/16,8,4, HN);
  _Float16* qa = (_Float16*)alloc((size_t)Mp*128*2);
  _Float16* qq = (_Float16*)alloc((size_t)Mp*128*2);
  GEMM(2,true, qf16,128, Wt_q1,128, f32in(I_QB1), 1.f, nullptr,0, qa,128, Mp/16,8,4, Mp);
  GEMM(1,true, qa,128,   Wt_q2,128, f32in(I_QB2), 1.f, nullptr,0, qq,128, Mp/16,8,4, Mp);

  // --- stage E: attention per head (scores -> softmax -> attn@V), WMMA both GEMMs ---
  float* scores   = (float*)alloc((size_t)Mp*NP*4);
  _Float16* attn  = (_Float16*)alloc((size_t)Mp*NP*2);
  _Float16* vt    = (_Float16*)alloc((size_t)128*NP*2);
  _Float16* enh   = (_Float16*)alloc((size_t)4*Mp*128*2);
  const float scl = 0.08838834764831845f;  // 1/sqrt(128)
  for (int h=0; h<4; ++h){
    GEMM(0,false, qq,128, kk + (size_t)h*NP*128,128, nullptr, scl,
         scores,NP, nullptr,0, Mp/16, NP/16, 4, Mp);
    softmax_kernel<<<(Mp*32+255)/256,256,0,stream>>>(scores, attn, Mp, N, NP);
    vtrans_kernel<<<(unsigned)(((long)128*NP+255)/256),256,0,stream>>>(vv + (size_t)h*NP*128, vt, NP);
    GEMM(0,false, attn,NP, vt,NP, nullptr, 1.f,
         nullptr,0, enh + (size_t)h*Mp*128,128, Mp/16, 8, NP/32, Mp);
  }

  // --- stage F: output head ---
  int HM = 4*Mp;
  _Float16* poa = (_Float16*)alloc((size_t)HM*128*2);
  _Float16* hoB = (_Float16*)alloc((size_t)HM*128*2);
  GEMM(2,true, enh,128, Wt_po1,128, f32in(I_POB1), 1.f, nullptr,0, poa,128, HM/16,8,4, HM);
  GEMM(1,true, poa,128, Wt_po2,128, f32in(I_POB2), 1.f, nullptr,0, hoB,128, HM/16,8,4, HM);
  _Float16* hoR = (_Float16*)alloc((size_t)Mp*512*2);
  hor_kernel<<<(unsigned)(((long)Mp*512+255)/256),256,0,stream>>>(hoB, hoR, Mp);
  _Float16* tf1 = (_Float16*)alloc((size_t)Mp*128*2);
  GEMM(2,true, hoR,512, Wt_f1,512, f32in(I_FB1), 1.f, nullptr,0, tf1,128, Mp/16,8,16, Mp);
  GEMM(2,true, tf1,128, Wt_f2,128, f32in(I_FB2), 1.f, (float*)d_out,128, nullptr,0, Mp/16,8,4, M);
}
